// CLCE_76493367542062
// MI455X (gfx1250) — compile-verified, matrix operand order checked by reading, not run on hardware
//
#include <hip/hip_runtime.h>
#include <math.h>

// Problem constants (from reference): N=4096, D=1024, C=512, TAU=0.5, LAMBD=0.5
#define N_ROWS 4096
#define DIMS   1024
#define NCLS   512

typedef __attribute__((ext_vector_type(16))) __bf16        v16bf;
typedef __attribute__((ext_vector_type(8)))  float         v8f;
typedef __attribute__((ext_vector_type(4)))  unsigned int  v4u;

union FragBF { v16bf v; v4u u[2]; };

__device__ __forceinline__ unsigned short f2bf_rne(float f) {
  unsigned int u = __float_as_uint(f);
  unsigned int r = u + 0x7FFFu + ((u >> 16) & 1u);
  return (unsigned short)(r >> 16);
}
__device__ __forceinline__ float bf2f(unsigned short s) {
  return __uint_as_float(((unsigned int)s) << 16);
}

// ---------------------------------------------------------------------------
// Kernel 1: row L2-normalize, cast to bf16 (RNE). One block per row.
// ---------------------------------------------------------------------------
__global__ void normalize_bf16_kernel(const float* __restrict__ x,
                                      unsigned short* __restrict__ xn) {
  __shared__ float red[256];
  const int row = blockIdx.x;
  const float* xr = x + (size_t)row * DIMS;
  float ss = 0.f;
#pragma unroll
  for (int k = 0; k < DIMS / 256; ++k) {
    float v = xr[threadIdx.x + k * 256];
    ss += v * v;
  }
  red[threadIdx.x] = ss;
  __syncthreads();
  for (int s = 128; s > 0; s >>= 1) {
    if (threadIdx.x < s) red[threadIdx.x] += red[threadIdx.x + s];
    __syncthreads();
  }
  const float inv = 1.0f / fmaxf(sqrtf(red[0]), 1e-8f);
  unsigned short* o = xn + (size_t)row * DIMS;
#pragma unroll
  for (int k = 0; k < DIMS / 256; ++k) {
    const int i = threadIdx.x + k * 256;
    o[i] = f2bf_rne(xr[i] * inv);
  }
}

// ---------------------------------------------------------------------------
// Kernel 2: per row: first same-class index (!= self) and #negatives.
// ---------------------------------------------------------------------------
__global__ void label_scan_kernel(const int* __restrict__ y,
                                  int* __restrict__ firstpos,
                                  int* __restrict__ nneg) {
  __shared__ int ly[N_ROWS];
  for (int i = threadIdx.x; i < N_ROWS; i += blockDim.x) ly[i] = y[i];
  __syncthreads();
  const int row = blockIdx.x * blockDim.x + threadIdx.x;
  const int my = ly[row];
  int same = 0, fp = -1;
  for (int j = 0; j < N_ROWS; ++j) {
    const bool s = (ly[j] == my);
    same += s ? 1 : 0;
    if (s && j != row && fp < 0) fp = j;
  }
  firstpos[row] = fp;
  nneg[row] = N_ROWS - same;  // 'same' includes self
}

// ---------------------------------------------------------------------------
// Kernel 3: slot0[i] = has_pos ? (xn_i . xn_firstpos + 1)*0.25 : 0.
// ---------------------------------------------------------------------------
__global__ void slot0_kernel(const unsigned short* __restrict__ xn,
                             const int* __restrict__ firstpos,
                             float* __restrict__ slot0) {
  __shared__ float red[256];
  const int row = blockIdx.x;
  const int p = firstpos[row];
  if (p < 0) {
    if (threadIdx.x == 0) slot0[row] = 0.f;
    return;
  }
  const unsigned short* a = xn + (size_t)row * DIMS;
  const unsigned short* b = xn + (size_t)p * DIMS;
  float d = 0.f;
#pragma unroll
  for (int k = 0; k < DIMS / 256; ++k) {
    const int i = threadIdx.x + k * 256;
    d += bf2f(a[i]) * bf2f(b[i]);
  }
  red[threadIdx.x] = d;
  __syncthreads();
  for (int s = 128; s > 0; s >>= 1) {
    if (threadIdx.x < s) red[threadIdx.x] += red[threadIdx.x + s];
    __syncthreads();
  }
  if (threadIdx.x == 0) slot0[row] = (red[0] + 1.0f) * 0.25f;
}

// ---------------------------------------------------------------------------
// Kernel 4: fused Gram + mask + exp row-sum.
//   partial[chunk][i] = sum over cols in chunk with y[col]!=y[i] of
//                       exp((xn_i . xn_col + 1)*0.25)
// Block = 256 threads = 8 waves, tile 64 rows x 128 cols per iteration,
// wave grid 2x4, each wave 32x32 = 2x2 WMMA 16x16 tiles, K=1024.
// Ping-pong fragment buffers with a 2-slab loop body: no register rotation
// moves, no spills, pressure < 256 VGPRs.
// ---------------------------------------------------------------------------
#define BM 64
#define BN 128
#define COLS_PER_CHUNK 512

#define LOAD_FRAGS(Fa0, Fa1, Fb0, Fb1, J)                                     \
  do {                                                                        \
    Fa0.u[0] = a0base[(J) + e0]; Fa0.u[1] = a0base[(J) + e1];                 \
    Fa1.u[0] = a1base[(J) + e0]; Fa1.u[1] = a1base[(J) + e1];                 \
    Fb0.u[0] = b0base[(J) + e0]; Fb0.u[1] = b0base[(J) + e1];                 \
    Fb1.u[0] = b1base[(J) + e0]; Fb1.u[1] = b1base[(J) + e1];                 \
  } while (0)

#define WMMA4(Fa0, Fa1, Fb0, Fb1)                                             \
  do {                                                                        \
    c00 = __builtin_amdgcn_wmma_f32_16x16x32_bf16(false, Fa0.v, false, Fb0.v, \
                                                  (short)0, c00, false, false);\
    c01 = __builtin_amdgcn_wmma_f32_16x16x32_bf16(false, Fa0.v, false, Fb1.v, \
                                                  (short)0, c01, false, false);\
    c10 = __builtin_amdgcn_wmma_f32_16x16x32_bf16(false, Fa1.v, false, Fb0.v, \
                                                  (short)0, c10, false, false);\
    c11 = __builtin_amdgcn_wmma_f32_16x16x32_bf16(false, Fa1.v, false, Fb1.v, \
                                                  (short)0, c11, false, false);\
  } while (0)

__global__ void __launch_bounds__(256) gram_exp_kernel(
    const unsigned short* __restrict__ xn,
    const int* __restrict__ y,
    float* __restrict__ partial) {
  __shared__ int yRow[BM];
  __shared__ int yCol[BN];
  __shared__ float sums[4][BM];

  const int tid = threadIdx.x;
  const int wid = tid >> 5;     // 0..7
  const int lane = tid & 31;
  const int l15 = lane & 15;
  const int hi = lane >> 4;

  const int waveR = wid >> 2;   // 0..1 : which 32-row half of the 64-row tile
  const int waveC = wid & 3;    // 0..3 : which 32-col quarter of the 128-col tile

  const int rowBlock = blockIdx.x * BM;
  const int chunkBase = blockIdx.y * COLS_PER_CHUNK;

  if (tid < BM) yRow[tid] = y[rowBlock + tid];

  // Per-lane accumulators for the wave's two 16-row subtiles.
  float accR0[8], accR1[8];
#pragma unroll
  for (int v = 0; v < 8; ++v) { accR0[v] = 0.f; accR1[v] = 0.f; }

  const int r0 = rowBlock + waveR * 32;
  const v4u* a0base = (const v4u*)(xn + (uint64_t)(r0 + l15) * DIMS);
  const v4u* a1base = (const v4u*)(xn + (uint64_t)(r0 + 16 + l15) * DIMS);

  // ISA 16-bit A/B layout: lane = M%16 (resp. N%16); lanes 0-15 hold
  // K {0..7,16..23}, lanes 16-31 hold K {8..15,24..31} -> two 16B loads.
  const int e0 = hi;        // v4u index offset for first 16B of a K-slab
  const int e1 = 2 + hi;    // v4u index offset for second 16B

  for (int it = 0; it < COLS_PER_CHUNK / BN; ++it) {
    const int colBase = chunkBase + it * BN;
    __syncthreads();  // protect yCol across iterations
    if (tid < BN) yCol[tid] = y[colBase + tid];
    __syncthreads();

    const int c0 = colBase + waveC * 32;
    const v4u* b0base = (const v4u*)(xn + (uint64_t)(c0 + l15) * DIMS);
    const v4u* b1base = (const v4u*)(xn + (uint64_t)(c0 + 16 + l15) * DIMS);

    v8f c00 = {}, c01 = {}, c10 = {}, c11 = {};

    FragBF a0c, a1c, b0c, b1c;   // even K-slab buffer
    FragBF a0n, a1n, b0n, b1n;   // odd  K-slab buffer

    LOAD_FRAGS(a0c, a1c, b0c, b1c, 0);  // slab k=0

#pragma unroll 1
    for (int k0 = 0; k0 < DIMS - 64; k0 += 64) {
      // Issue odd-slab loads, overlap with WMMAs on the even slab.
      LOAD_FRAGS(a0n, a1n, b0n, b1n, (k0 + 32) >> 3);
      __builtin_prefetch((const void*)(a0base + ((k0 + 64) >> 3)), 0, 1);
      __builtin_prefetch((const void*)(b0base + ((k0 + 64) >> 3)), 0, 1);
      WMMA4(a0c, a1c, b0c, b1c);
      // Refill the even buffer, overlap with WMMAs on the odd slab.
      LOAD_FRAGS(a0c, a1c, b0c, b1c, (k0 + 64) >> 3);
      WMMA4(a0n, a1n, b0n, b1n);
    }
    // Peeled tail: slabs DIMS-64 (already in cur) and DIMS-32.
    LOAD_FRAGS(a0n, a1n, b0n, b1n, (DIMS - 32) >> 3);
    WMMA4(a0c, a1c, b0c, b1c);
    WMMA4(a0n, a1n, b0n, b1n);

    // C layout: lanes 0-15: (M=v, N=lane); lanes 16-31: (M=v+8, N=lane-16).
    const int mOff = hi * 8;
    const int ycl0 = yCol[waveC * 32 + l15];
    const int ycl1 = yCol[waveC * 32 + 16 + l15];
#pragma unroll
    for (int v = 0; v < 8; ++v) {
      const int yr0 = yRow[waveR * 32 + mOff + v];
      const int yr1 = yRow[waveR * 32 + 16 + mOff + v];
      float s;
      s = (c00[v] + 1.0f) * 0.25f; if (ycl0 != yr0) accR0[v] += __expf(s);
      s = (c01[v] + 1.0f) * 0.25f; if (ycl1 != yr0) accR0[v] += __expf(s);
      s = (c10[v] + 1.0f) * 0.25f; if (ycl0 != yr1) accR1[v] += __expf(s);
      s = (c11[v] + 1.0f) * 0.25f; if (ycl1 != yr1) accR1[v] += __expf(s);
    }
  }

  // Fixed-order XOR reduction across the 16 column-lanes of each half-wave.
#pragma unroll
  for (int v = 0; v < 8; ++v) {
#pragma unroll
    for (int m = 1; m < 16; m <<= 1) {
      accR0[v] += __shfl_xor(accR0[v], m, 32);
      accR1[v] += __shfl_xor(accR1[v], m, 32);
    }
  }
  if (l15 == 0) {  // lanes 0 and 16 hold totals for their 8-row sets
    const int base = waveR * 32 + hi * 8;
#pragma unroll
    for (int v = 0; v < 8; ++v) {
      sums[waveC][base + v]      = accR0[v];
      sums[waveC][base + 16 + v] = accR1[v];
    }
  }
  __syncthreads();
  if (tid < BM) {
    const float S = sums[0][tid] + sums[1][tid] + sums[2][tid] + sums[3][tid];
    partial[(size_t)blockIdx.y * N_ROWS + rowBlock + tid] = S;
  }
}

// ---------------------------------------------------------------------------
// Kernel 5: cross-entropy per row of y_pred. One block per row.
// ---------------------------------------------------------------------------
__global__ void ce_kernel(const float* __restrict__ yp,
                          const int* __restrict__ y,
                          float* __restrict__ ce_rows) {
  __shared__ float red[256];
  const int row = blockIdx.x;
  const float* r = yp + (size_t)row * NCLS;
  float m = -3.4e38f;
#pragma unroll
  for (int k = 0; k < NCLS / 256; ++k) m = fmaxf(m, r[threadIdx.x + k * 256]);
  red[threadIdx.x] = m;
  __syncthreads();
  for (int s = 128; s > 0; s >>= 1) {
    if (threadIdx.x < s) red[threadIdx.x] = fmaxf(red[threadIdx.x], red[threadIdx.x + s]);
    __syncthreads();
  }
  m = red[0];
  __syncthreads();
  float sum = 0.f;
#pragma unroll
  for (int k = 0; k < NCLS / 256; ++k) sum += __expf(r[threadIdx.x + k * 256] - m);
  red[threadIdx.x] = sum;
  __syncthreads();
  for (int s = 128; s > 0; s >>= 1) {
    if (threadIdx.x < s) red[threadIdx.x] += red[threadIdx.x + s];
    __syncthreads();
  }
  if (threadIdx.x == 0) {
    const float lse = m + __logf(red[0]);
    ce_rows[row] = lse - r[y[row]];
  }
}

// ---------------------------------------------------------------------------
// Kernel 6: cl_i = log(e^{slot0} + S_neg + #zeros) - slot0
//   (#zeros = 2N-2 - nneg; reference row is [slot0, negs..., zeros] len 2N-1)
// ---------------------------------------------------------------------------
__global__ void cl_rows_kernel(const float* __restrict__ partial,
                               const float* __restrict__ slot0,
                               const int* __restrict__ nneg,
                               float* __restrict__ cl_rows) {
  const int i = blockIdx.x * blockDim.x + threadIdx.x;
  float S = 0.f;
#pragma unroll
  for (int c = 0; c < 8; ++c) S += partial[(size_t)c * N_ROWS + i];
  const float Z = (float)(2 * N_ROWS - 2 - nneg[i]);
  const float s0 = slot0[i];
  cl_rows[i] = __logf(__expf(s0) + S + Z) - s0;
}

// ---------------------------------------------------------------------------
// Kernel 7: final deterministic reduction -> scalar loss.
// ---------------------------------------------------------------------------
__global__ void final_reduce_kernel(const float* __restrict__ cl_rows,
                                    const float* __restrict__ ce_rows,
                                    float* __restrict__ out) {
  __shared__ float rc[256], re[256];
  float a = 0.f, b = 0.f;
#pragma unroll
  for (int k = 0; k < N_ROWS / 256; ++k) {
    a += cl_rows[threadIdx.x + k * 256];
    b += ce_rows[threadIdx.x + k * 256];
  }
  rc[threadIdx.x] = a;
  re[threadIdx.x] = b;
  __syncthreads();
  for (int s = 128; s > 0; s >>= 1) {
    if (threadIdx.x < s) {
      rc[threadIdx.x] += rc[threadIdx.x + s];
      re[threadIdx.x] += re[threadIdx.x + s];
    }
    __syncthreads();
  }
  if (threadIdx.x == 0)
    out[0] = 0.5f * (rc[0] / (float)N_ROWS) + 0.5f * (re[0] / (float)N_ROWS);
}

// ---------------------------------------------------------------------------
extern "C" void kernel_launch(void* const* d_in, const int* in_sizes, int n_in,
                              void* d_out, int out_size, void* d_ws, size_t ws_size,
                              hipStream_t stream) {
  const float* x  = (const float*)d_in[0];   // (4096, 1024) f32
  const int*   y  = (const int*)d_in[1];     // (4096,) i32
  const float* yp = (const float*)d_in[2];   // (4096, 512) f32
  float* out = (float*)d_out;

  char* ws = (char*)d_ws;
  size_t off = 0;
  unsigned short* xn = (unsigned short*)(ws + off); off += (size_t)N_ROWS * DIMS * 2;  // 8 MB
  float* partial  = (float*)(ws + off); off += 8 * (size_t)N_ROWS * sizeof(float);
  float* slot0    = (float*)(ws + off); off += N_ROWS * sizeof(float);
  int*   firstpos = (int*)(ws + off);   off += N_ROWS * sizeof(int);
  int*   nneg     = (int*)(ws + off);   off += N_ROWS * sizeof(int);
  float* cl_rows  = (float*)(ws + off); off += N_ROWS * sizeof(float);
  float* ce_rows  = (float*)(ws + off); off += N_ROWS * sizeof(float);

  normalize_bf16_kernel<<<N_ROWS, 256, 0, stream>>>(x, xn);
  label_scan_kernel<<<N_ROWS / 256, 256, 0, stream>>>(y, firstpos, nneg);
  slot0_kernel<<<N_ROWS, 256, 0, stream>>>(xn, firstpos, slot0);
  gram_exp_kernel<<<dim3(N_ROWS / BM, N_ROWS / COLS_PER_CHUNK), 256, 0, stream>>>(xn, y, partial);
  ce_kernel<<<N_ROWS, 256, 0, stream>>>(yp, y, ce_rows);
  cl_rows_kernel<<<N_ROWS / 256, 256, 0, stream>>>(partial, slot0, nneg, cl_rows);
  final_reduce_kernel<<<1, 256, 0, stream>>>(cl_rows, ce_rows, out);
}
